// Tucker3D_89240830476843
// MI455X (gfx1250) — matrix operand order repaired and checked by reading, not compiled, not run on hardware
//
#include <hip/hip_runtime.h>

typedef __attribute__((ext_vector_type(16))) _Float16 v16h;
typedef __attribute__((ext_vector_type(8)))  float    v8f;
typedef int v4i __attribute__((vector_size(16)));

#define WAVES          8
#define PTS_PER_WAVE   16
#define PTS_PER_BLOCK  (WAVES * PTS_PER_WAVE)   // 128
#define PLANE_HW       65536                    // 256*256
#define LDS_GSW        65536                    // 32*32*32 f16
#define LDS_FXY        (WAVES * 16 * 32 * 2)    // 8192
#define LDS_FYZ        (WAVES * 16 * 32 * 4)    // 16384
#define LDS_BYTES      (LDS_GSW + LDS_FXY + 2 * LDS_FYZ)  // 106496

#define AS1 __attribute__((address_space(1)))
#define AS3 __attribute__((address_space(3)))

#if __has_builtin(__builtin_amdgcn_global_load_async_to_lds_b128)
#define HAVE_ASYNC_LDS 1
#endif

// ---------- plane transpose: (C=32,H,W) f32  ->  (H,W,C=32) f16 ----------
__global__ void plane_hwc_f16_kernel(const float* __restrict__ src,
                                     _Float16* __restrict__ dst) {
    int pix = blockIdx.x * blockDim.x + threadIdx.x;     // 0..65535 = y*256+x
    if (pix >= PLANE_HW) return;
    _Float16 tmp[32];
#pragma unroll
    for (int c = 0; c < 32; ++c)
        tmp[c] = (_Float16)src[(size_t)c * PLANE_HW + pix];
    uint4* d = (uint4*)(dst + (size_t)pix * 32);
    const uint4* s = (const uint4*)tmp;
    d[0] = s[0]; d[1] = s[1]; d[2] = s[2]; d[3] = s[3];
}

// ---------- G (f32, x*1024+y*32+z) -> f16 in WMMA-B swizzled layout ----------
// tile t covers columns j=16t..16t+15; lane L<16: n=L%16, K=2v+lo; L>=16: K=16+2v+lo
__global__ void g_swizzle_kernel(const float* __restrict__ G,
                                 _Float16* __restrict__ Gsw) {
    int e = blockIdx.x * blockDim.x + threadIdx.x;
    if (e >= 32768) return;
    int k = e >> 10;                 // x (K dim)
    int j = e & 1023;                // y*32+z (column)
    int t = j >> 4, cc = j & 15;
    int hi = k >> 4, kk = k & 15;
    Gsw[(t * 32 + (hi * 16 + cc)) * 16 + kk] = (_Float16)G[e];
}

// ---------- bilinear taps ----------
__device__ __forceinline__ float bisample_hwc(const _Float16* __restrict__ tp,
                                              int c, float gx, float gy) {
    float px = (gx + 1.0f) * 0.5f * 255.0f;
    float py = (gy + 1.0f) * 0.5f * 255.0f;
    float x0f = floorf(px), y0f = floorf(py);
    float wx1 = px - x0f, wy1 = py - y0f;
    float wx0 = 1.0f - wx1, wy0 = 1.0f - wy1;
    int x0 = (int)x0f, y0 = (int)y0f;
    float acc = 0.0f;
#pragma unroll
    for (int t = 0; t < 4; ++t) {
        int ix = x0 + (t & 1), iy = y0 + (t >> 1);
        float w = ((t & 1) ? wx1 : wx0) * ((t >> 1) ? wy1 : wy0);
        if ((unsigned)ix < 256u && (unsigned)iy < 256u)
            acc += (float)tp[(size_t)((iy << 8) + ix) * 32 + c] * w;
    }
    return acc;
}

__device__ __forceinline__ float bisample_chw(const float* __restrict__ plane,
                                              int c, float gx, float gy) {
    float px = (gx + 1.0f) * 0.5f * 255.0f;
    float py = (gy + 1.0f) * 0.5f * 255.0f;
    float x0f = floorf(px), y0f = floorf(py);
    float wx1 = px - x0f, wy1 = py - y0f;
    float wx0 = 1.0f - wx1, wy0 = 1.0f - wy1;
    int x0 = (int)x0f, y0 = (int)y0f;
    const float* b = plane + (size_t)c * PLANE_HW;
    float acc = 0.0f;
#pragma unroll
    for (int t = 0; t < 4; ++t) {
        int ix = x0 + (t & 1), iy = y0 + (t >> 1);
        float w = ((t & 1) ? wx1 : wx0) * ((t >> 1) ? wy1 : wy0);
        if ((unsigned)ix < 256u && (unsigned)iy < 256u)
            acc += b[(iy << 8) + ix] * w;
    }
    return acc;
}

// ---------- fused tri-plane sample + Tucker contraction (WMMA) ----------
__global__ void __launch_bounds__(PTS_PER_BLOCK * 2, 1)
tucker3d_kernel(const float* __restrict__ coords,      // (N,3)
                const float* __restrict__ pXY, const float* __restrict__ pYZ,
                const float* __restrict__ pXZ,
                const _Float16* __restrict__ tXY, const _Float16* __restrict__ tYZ,
                const _Float16* __restrict__ tXZ,
                const float* __restrict__ G,           // (32,32,32) f32
                const _Float16* __restrict__ GswG,     // pre-swizzled f16 (ws)
                float* __restrict__ out, int npts, float invR, int use_t) {
    extern __shared__ char smem[];
    _Float16* Gsw = (_Float16*)smem;                         // B-swizzled G, f16
    _Float16* fXY = (_Float16*)(smem + LDS_GSW);             // [WAVES][16][32]
    float*    fYZ = (float*)(smem + LDS_GSW + LDS_FXY);
    float*    fXZ = (float*)(smem + LDS_GSW + LDS_FXY + LDS_FYZ);

    const int tid  = threadIdx.x;
    const int lane = tid & 31;
    const int wave = tid >> 5;

    // Phase 1: stage swizzled G into LDS.
    if (use_t) {
#ifdef HAVE_ASYNC_LDS
        // Async DMA: 64KB as 4096 x b128 chunks, 16 per thread. Overlaps phase 2.
#pragma unroll
        for (int i = 0; i < 16; ++i) {
            int q = tid + i * PTS_PER_BLOCK * 2;
            __builtin_amdgcn_global_load_async_to_lds_b128(
                (AS1 v4i*)((const char*)GswG + (size_t)q * 16),
                (AS3 v4i*)(smem + (size_t)q * 16), 0, 0);
        }
#else
        for (int q = tid; q < LDS_GSW / 16; q += blockDim.x)
            ((uint4*)smem)[q] = ((const uint4*)GswG)[q];
#endif
    } else {
        // No workspace: convert + swizzle in-kernel.
        for (int e = tid; e < 32768; e += blockDim.x) {
            int k = e >> 10, j = e & 1023;
            int t = j >> 4, cc = j & 15;
            int hi = k >> 4, kk = k & 15;
            Gsw[(t * 32 + (hi * 16 + cc)) * 16 + kk] = (_Float16)G[e];
        }
    }

    // Phase 2: per-wave feature sampling (16 points; lane = channel)
    const int pbase = blockIdx.x * PTS_PER_BLOCK + wave * PTS_PER_WAVE;
    if (use_t) {
        for (int p = 0; p < PTS_PER_WAVE; ++p) {
            int n = pbase + p;
            if (n >= npts) n = npts - 1;
            float cx = coords[3 * n + 0] * invR;
            float cy = coords[3 * n + 1] * invR;
            float cz = coords[3 * n + 2] * invR;
            int fb = (wave * 16 + p) * 32 + lane;
            fXY[fb] = (_Float16)bisample_hwc(tXY, lane, cx, cy);
            fYZ[fb] = bisample_hwc(tYZ, lane, cy, cz);
            fXZ[fb] = bisample_hwc(tXZ, lane, cx, cz);
        }
    } else {
        for (int p = 0; p < PTS_PER_WAVE; ++p) {
            int n = pbase + p;
            if (n >= npts) n = npts - 1;
            float cx = coords[3 * n + 0] * invR;
            float cy = coords[3 * n + 1] * invR;
            float cz = coords[3 * n + 2] * invR;
            int fb = (wave * 16 + p) * 32 + lane;
            fXY[fb] = (_Float16)bisample_chw(pXY, lane, cx, cy);
            fYZ[fb] = bisample_chw(pYZ, lane, cy, cz);
            fXZ[fb] = bisample_chw(pXZ, lane, cx, cz);
        }
    }

#ifdef HAVE_ASYNC_LDS
    if (use_t) asm volatile("s_wait_asynccnt 0x0" ::: "memory");
#endif
    __syncthreads();

    // Phase 3: assemble A (16 pts x 32 K, f16) in contiguous-K layout (K = 16*hi + element)
    const int pr = lane & 15;      // row (lanes 0-15 and 16-31 both hold M=0..15)
    const int hi = lane >> 4;      // K-half select
    union { v16h h; unsigned int u[8]; } A;
#pragma unroll
    for (int v = 0; v < 8; ++v) {
        int K = (hi << 4) + (v << 1);
        A.u[v] = *(const unsigned int*)&fXY[(wave * 16 + pr) * 32 + K];
    }

    float xz0[8], xz1[8], acc[8];
#pragma unroll
    for (int r = 0; r < 8; ++r) {
        int m = r + (hi << 3);
        xz0[r] = fXZ[(wave * 16 + m) * 32 + pr];        // z = pr      (even tiles)
        xz1[r] = fXZ[(wave * 16 + m) * 32 + 16 + pr];   // z = 16 + pr (odd tiles)
        acc[r] = 0.0f;
    }

    // Phase 4: 64 WMMA tiles processed as 32 pairs sharing y:
    //   acc[r] += yz[y] * (D_even[r]*xz0[r] + D_odd[r]*xz1[r])
    // One-stage pipeline over pairs: fold pair u while pair u+1's two WMMAs are
    // in flight, so WMMA->VALU hazard slots are covered by the fold's own
    // independent mul/fma chains. 3 VALU ops per row per pair (vs 4 before).
    auto loadB = [&](int t) -> v16h {
        union { v16h h; uint4 q[2]; } B;
        const uint4* gq = (const uint4*)(Gsw + (t * 32 + lane) * 16);
        B.q[0] = gq[0];
        B.q[1] = gq[1];
        return B.h;
    };

    v8f czero = {};
    v8f de = __builtin_amdgcn_wmma_f32_16x16x32_f16(
        false, A.h, false, loadB(0), (short)0, czero, false, false);
    v8f dodd = __builtin_amdgcn_wmma_f32_16x16x32_f16(
        false, A.h, false, loadB(1), (short)0, czero, false, false);
#pragma unroll 4
    for (int u = 0; u < 32; ++u) {
        float yzv[8];
#pragma unroll
        for (int r = 0; r < 8; ++r)
            yzv[r] = fYZ[(wave * 16 + r + (hi << 3)) * 32 + u];
        v8f den = __builtin_amdgcn_wmma_f32_16x16x32_f16(
            false, A.h, false, loadB((2 * u + 2) & 63), (short)0, czero, false, false);
        v8f don = __builtin_amdgcn_wmma_f32_16x16x32_f16(
            false, A.h, false, loadB((2 * u + 3) & 63), (short)0, czero, false, false);
#pragma unroll
        for (int r = 0; r < 8; ++r) {
            float s = de[r] * xz0[r];
            s = fmaf(dodd[r], xz1[r], s);
            acc[r] = fmaf(yzv[r], s, acc[r]);
        }
        de = den;
        dodd = don;
    }

    // Phase 5: sum over the 16 column-lanes, store (lane 0 -> pt r, lane 16 -> pt 8+r)
#pragma unroll
    for (int r = 0; r < 8; ++r) {
        float v = acc[r];
        v += __shfl_xor(v, 1, 32);
        v += __shfl_xor(v, 2, 32);
        v += __shfl_xor(v, 4, 32);
        v += __shfl_xor(v, 8, 32);
        int n = pbase + (hi << 3) + r;
        if (pr == 0 && n < npts) out[n] = v;
    }
}

extern "C" void kernel_launch(void* const* d_in, const int* in_sizes, int n_in,
                              void* d_out, int out_size, void* d_ws, size_t ws_size,
                              hipStream_t stream) {
    const float* coords = (const float*)d_in[0];
    const float* pXY    = (const float*)d_in[1];
    const float* pYZ    = (const float*)d_in[2];
    const float* pXZ    = (const float*)d_in[3];
    const float* G      = (const float*)d_in[4];
    float* out = (float*)d_out;

    int npts = in_sizes[0] / 3;
    const size_t planeBytes = (size_t)PLANE_HW * 32 * sizeof(_Float16); // 4 MB
    const size_t gswBytes   = 32768 * sizeof(_Float16);                 // 64 KB
    int use_t = (ws_size >= 3 * planeBytes + gswBytes) ? 1 : 0;

    _Float16* tXY  = (_Float16*)d_ws;
    _Float16* tYZ  = (_Float16*)((char*)d_ws + planeBytes);
    _Float16* tXZ  = (_Float16*)((char*)d_ws + 2 * planeBytes);
    _Float16* GswG = (_Float16*)((char*)d_ws + 3 * planeBytes);

    if (use_t) {
        dim3 tb(256), tg(PLANE_HW / 256);
        plane_hwc_f16_kernel<<<tg, tb, 0, stream>>>(pXY, tXY);
        plane_hwc_f16_kernel<<<tg, tb, 0, stream>>>(pYZ, tYZ);
        plane_hwc_f16_kernel<<<tg, tb, 0, stream>>>(pXZ, tXZ);
        g_swizzle_kernel<<<dim3(128), dim3(256), 0, stream>>>(G, GswG);
    }

    int blocks = (npts + PTS_PER_BLOCK - 1) / PTS_PER_BLOCK;  // 2048 for N=262144
    tucker3d_kernel<<<blocks, PTS_PER_BLOCK * 2, LDS_BYTES, stream>>>(
        coords, pXY, pYZ, pXZ, tXY, tYZ, tXZ, G, GswG, out, npts, 1.0f / 1.3f, use_t);
}
